// HungarianMatcher_58110907515474
// MI455X (gfx1250) — matrix operand order repaired and checked by reading, not compiled, not run on hardware
//
#include <hip/hip_runtime.h>
#include <hip/hip_bf16.h>

typedef __attribute__((ext_vector_type(16))) _Float16 v16h;
typedef __attribute__((ext_vector_type(8)))  _Float16 v8h;
typedef __attribute__((ext_vector_type(8)))  float    v8f;
typedef __attribute__((ext_vector_type(8)))  unsigned v8u;

#define EPSB 1e-6f
#define BIGC 1000000.0f

// nan_to_num(logits, nan=0, posinf=0, neginf=0): any non-finite -> 0
__device__ __forceinline__ float san_logit(float x) {
    return __builtin_isfinite(x) ? x : 0.0f;
}

// nan_to_num(box, nan=0, posinf=1, neginf=0)
__device__ __forceinline__ float san_box(float x) {
    if (__builtin_isnan(x)) return 0.0f;
    if (__builtin_isinf(x)) return x > 0.0f ? 1.0f : 0.0f;
    return x;
}

__device__ __forceinline__ void prep_box(float4 r, float& cx, float& cy, float& w, float& h) {
    cx = fminf(fmaxf(san_box(r.x), 0.0f), 1.0f);
    cy = fminf(fmaxf(san_box(r.y), 0.0f), 1.0f);
    w  = fminf(fmaxf(san_box(r.z), EPSB), 1.0f);
    h  = fminf(fmaxf(san_box(r.w), EPSB), 1.0f);
}

// ---------------------------------------------------------------------------
// Kernel 1: row softmax over C=256, store f16 probs row-major [B*Q, 256].
// One wave per row; 8 f32 per lane; wave32 shuffle reductions.
// ---------------------------------------------------------------------------
__global__ __launch_bounds__(256) void softmax_rows_f16(
    const float* __restrict__ logits, _Float16* __restrict__ probs, int nrows)
{
    int row = blockIdx.x * 8 + (threadIdx.x >> 5);
    if (row >= nrows) return;                 // wave-uniform
    int lane = threadIdx.x & 31;

    const float* rp = logits + (size_t)row * 256 + lane * 8;
    float v[8];
    float4 a = *(const float4*)rp;
    float4 b = *(const float4*)(rp + 4);
    v[0] = a.x; v[1] = a.y; v[2] = a.z; v[3] = a.w;
    v[4] = b.x; v[5] = b.y; v[6] = b.z; v[7] = b.w;
#pragma unroll
    for (int i = 0; i < 8; ++i) v[i] = san_logit(v[i]);

    float m = v[0];
#pragma unroll
    for (int i = 1; i < 8; ++i) m = fmaxf(m, v[i]);
#pragma unroll
    for (int s = 1; s < 32; s <<= 1) m = fmaxf(m, __shfl_xor(m, s, 32));

    float sum = 0.0f;
#pragma unroll
    for (int i = 0; i < 8; ++i) { v[i] = __expf(v[i] - m); sum += v[i]; }
#pragma unroll
    for (int s = 1; s < 32; s <<= 1) sum += __shfl_xor(sum, s, 32);

    float inv = 1.0f / sum;
    v8h o;
#pragma unroll
    for (int i = 0; i < 8; ++i) o[i] = (_Float16)(v[i] * inv);
    *(v8h*)(probs + (size_t)row * 256 + lane * 8) = o;
}

// ---------------------------------------------------------------------------
// Kernel 2: one wave computes a 64(q) x 16(t) tile of the cost matrix.
//   A (M=16 t-rows x K=32 c) = one-hot(labels), built ONCE per wave for all
//       8 K-steps (64 VGPRs), reused across 4 q-subtiles.
//   B (K=32 c x N=16 q-cols) = f16 softmax probs; per lane one contiguous
//       32B load per K-step.
//   D rows = t, cols = q  ->  per lane: fixed q, 8 consecutive t values
//       -> L1/GIoU fused, output stored as two b128 per lane per subtile.
// ---------------------------------------------------------------------------
__global__ __launch_bounds__(32) void cost_tile_wmma(
    const _Float16* __restrict__ probs,   // [B*Q, 256] f16
    const float*    __restrict__ pboxes,  // [B*Q, 4]
    const int*      __restrict__ labels,  // [B, T]
    const float*    __restrict__ tboxes,  // [B*T, 4]
    float*          __restrict__ out,     // [B, Q, T]
    int Q, int T)
{
    const int b     = blockIdx.z;
    const int qbase = blockIdx.x * 64;
    const int tbase = blockIdx.y * 16;
    const int lane  = threadIdx.x;
    const int half  = lane >> 4;          // 0: lanes 0-15, 1: lanes 16-31
    const int ln    = lane & 15;

    // ================= one-hot A, built once for all 8 K-steps ============
    // A-matrix 16x32 f16 layout: lane holds row M=ln; halves 0..15 -> K =
    // kbase+aoff+0..15, aoff = half*16.
    {
    }
    int tla = tbase + ln; tla = tla < T ? tla : T - 1;
    const int lab     = labels[b * T + tla];
    const int aoff    = half * 16;
    const int relbase = lab - aoff;                         // K-offset of the 1
    const unsigned vv = (relbase & 1) ? 0x3C000000u : 0x00003C00u; // f16 1.0
    const int selbase = relbase >> 1;                       // word selector

    v16h Aoh[8];
#pragma unroll
    for (int kk = 0; kk < 8; ++kk) {
        const int sel = selbase - kk * 16;                  // (rel>>1) this step
        v8u w;
#pragma unroll
        for (int i = 0; i < 8; ++i)
            w[i] = (sel == i) ? vv : 0u;  // words 0..7: K+0..15
        Aoh[kk] = __builtin_bit_cast(v16h, w);
    }

    // ============ target boxes: 8 t-rows per lane, hoisted per wave =======
    float tcx[8], tcy[8], twv[8], thv[8];
    float tx0[8], ty0[8], tx1[8], ty1[8], ta[8];
#pragma unroll
    for (int r = 0; r < 8; ++r) {
        int tg = tbase + r + 8 * half; int tc = tg < T ? tg : T - 1;
        prep_box(((const float4*)tboxes)[(size_t)b * T + tc],
                 tcx[r], tcy[r], twv[r], thv[r]);
        tx0[r] = tcx[r] - 0.5f * twv[r]; ty0[r] = tcy[r] - 0.5f * thv[r];
        tx1[r] = tcx[r] + 0.5f * twv[r]; ty1[r] = tcy[r] + 0.5f * thv[r];
        ta[r]  = twv[r] * thv[r];
    }
    const int koff = half * 16;   // B-matrix K half for this lane

    // ======================= 4 q-subtiles of 16 ===========================
#pragma unroll
    for (int sub = 0; sub < 4; ++sub) {
        const int qg = qbase + sub * 16 + ln;               // this lane's q col
        const int qc = qg < Q ? qg : Q - 1;
        const _Float16* brow = probs + ((size_t)b * Q + qc) * 256 + koff;

        v8f acc = {};
#pragma unroll
        for (int kk = 0; kk < 8; ++kk) {
            // B: lane = column N=q, 16 contiguous halves at K = kk*32+koff
            v16h Bv = *(const v16h*)(brow + kk * 32);
            acc = __builtin_amdgcn_wmma_f32_16x16x32_f16(
                false, Aoh[kk], false, Bv, (short)0, acc, false, false);
        }

        // pred box: one per lane (its q column)
        float pcx, pcy, pw, ph;
        prep_box(((const float4*)pboxes)[(size_t)b * Q + qc], pcx, pcy, pw, ph);
        const float px0 = pcx - 0.5f * pw, py0 = pcy - 0.5f * ph;
        const float px1 = pcx + 0.5f * pw, py1 = pcy + 0.5f * ph;
        const float pa  = pw * ph;

        float cost[8];
#pragma unroll
        for (int r = 0; r < 8; ++r) {
            const float l1 = fabsf(pcx - tcx[r]) + fabsf(pcy - tcy[r])
                           + fabsf(pw  - twv[r]) + fabsf(ph  - thv[r]);

            const float iw    = fmaxf(fminf(px1, tx1[r]) - fmaxf(px0, tx0[r]), 0.0f);
            const float ih    = fmaxf(fminf(py1, ty1[r]) - fmaxf(py0, ty0[r]), 0.0f);
            const float inter = iw * ih;
            const float uni   = pa + ta[r] - inter;
            const float iou   = inter / uni;
            const float cw    = fmaxf(px1, tx1[r]) - fminf(px0, tx0[r]);
            const float ch    = fmaxf(py1, ty1[r]) - fminf(py0, ty0[r]);
            const float ac    = cw * ch;
            const float giou  = iou - (ac - uni) / ac;

            float c = -acc[r] + 5.0f * l1 - 2.0f * giou;
            cost[r] = __builtin_isfinite(c) ? c : BIGC;
        }

        // store: lane's 8 costs are t-contiguous at t0 = tbase + 8*half
        const int t0 = tbase + 8 * half;
        if (qg < Q) {
            float* op = out + ((size_t)b * Q + qg) * T + t0;
            if (t0 + 7 < T) {                      // full 8-wide (T % 8 == 0)
                float4 lo4 = make_float4(cost[0], cost[1], cost[2], cost[3]);
                float4 hi4 = make_float4(cost[4], cost[5], cost[6], cost[7]);
                *(float4*)(op)     = lo4;
                *(float4*)(op + 4) = hi4;
            } else {
#pragma unroll
                for (int r = 0; r < 8; ++r)
                    if (t0 + r < T) op[r] = cost[r];
            }
        }
    }
}

extern "C" void kernel_launch(void* const* d_in, const int* in_sizes, int n_in,
                              void* d_out, int out_size, void* d_ws, size_t ws_size,
                              hipStream_t stream) {
    (void)in_sizes; (void)n_in; (void)out_size; (void)ws_size;
    const float* logits = (const float*)d_in[0];   // [B,Q,C] f32
    const float* pboxes = (const float*)d_in[1];   // [B,Q,4] f32
    const int*   labels = (const int*)  d_in[2];   // [B,T]   i32
    const float* tboxes = (const float*)d_in[3];   // [B,T,4] f32
    float* out = (float*)d_out;                    // [B,Q,T] f32

    constexpr int B = 64, Q = 900, T = 200;        // C = 256 baked into kernels
    _Float16* probs = (_Float16*)d_ws;             // needs B*Q*256*2 = ~29.5 MB

    const int nrows = B * Q;
    softmax_rows_f16<<<(nrows + 7) / 8, 256, 0, stream>>>(logits, probs, nrows);

    dim3 grid((Q + 63) / 64, (T + 15) / 16, B);    // 15 x 13 x 64 tiles
    cost_tile_wmma<<<grid, 32, 0, stream>>>(probs, pboxes, labels, tboxes, out, Q, T);
}
